// MultiHeadSelfAttention_83245056131913
// MI455X (gfx1250) — compile-verified, matrix operand order checked by reading, not run on hardware
//
#include <hip/hip_runtime.h>
#include <hip/hip_bf16.h>
#include <math.h>

// ---------------------------------------------------------------------------
// Types / WMMA helpers (gfx1250, wave32)
// ---------------------------------------------------------------------------
typedef __bf16 bf16_t;
typedef __attribute__((ext_vector_type(16))) __bf16 bf16x16;
typedef __attribute__((ext_vector_type(8)))  __bf16 bf16x8;
typedef __attribute__((ext_vector_type(4)))  __bf16 bf16x4;
typedef __attribute__((ext_vector_type(8)))  float  f32x8;

union FragBF {
    bf16x16 v;
    struct { bf16x8 lo, hi; } h;
};

__device__ __forceinline__ bf16x16 load_frag16(const bf16_t* p0, const bf16_t* p1) {
    FragBF f;
    f.h.lo = *(const bf16x8*)p0;   // 16B: K offsets {0..7}+8*half
    f.h.hi = *(const bf16x8*)p1;   // 16B: K offsets {16..23}+8*half
    return f.v;
}

__device__ __forceinline__ f32x8 wmma_bf16(bf16x16 a, bf16x16 b, f32x8 c) {
    return __builtin_amdgcn_wmma_f32_16x16x32_bf16(
        false, a, false, b, (short)0, c, false, false);
}

__device__ __forceinline__ f32x8 zero8() {
    f32x8 z;
#pragma unroll
    for (int i = 0; i < 8; ++i) z[i] = 0.0f;
    return z;
}

// ---------------------------------------------------------------------------
// Problem constants
// ---------------------------------------------------------------------------
#define D_MODEL 1024
#define N_HEADS 16
#define D_HEAD  64
#define SEQ_T   2048
#define BATCH   2
#define M_TOT   (BATCH * SEQ_T)      // 4096

#define MBLK 256                     // 8 waves x 32 rows
#define NBLK 64                      // 4 n-tiles of 16
#define KSTEP 32
#define LDA (KSTEP + 8)              // 40 bf16 = 80B row stride (16B aligned)

// ===========================================================================
// Kernel 1: QKV projection  qkv = x @ W_qkv + b_qkv
//   Q scaled by 1/sqrt(dh) -> [B,H,T,dh]; K -> [B,H,T,dh]; V -> [B,H,dh,T].
// grid: (3072/NBLK, 4096/MBLK) = (48,16), block 256
// ===========================================================================
__global__ __launch_bounds__(256) void qkv_kernel(
    const float* __restrict__ x, const float* __restrict__ Wq,
    const float* __restrict__ bq,
    bf16_t* __restrict__ Q, bf16_t* __restrict__ K, bf16_t* __restrict__ Vt)
{
    __shared__ bf16_t As[2][MBLK * LDA];
    __shared__ bf16_t Ws[2][NBLK * LDA];
    const int n0   = blockIdx.x * NBLK;
    const int m0   = blockIdx.y * MBLK;
    const int tid  = threadIdx.x;
    const int lane = tid & 31, w = tid >> 5;
    const int half = lane >> 4, r = lane & 15;

    f32x8 acc[2][4];
#pragma unroll
    for (int a = 0; a < 2; ++a)
#pragma unroll
        for (int i = 0; i < 4; ++i) acc[a][i] = zero8();

    float4 ra[8];   // A tile: 256x32 f32 = 2048 float4 / 256 thr = 8 each
    float4 rw[2];   // W tile: 32x64  f32 =  512 float4 / 256 thr = 2 each

    // ---- prologue: stage tile 0 ----
#pragma unroll
    for (int i = 0; i < 8; ++i) {
        int f = tid + i * 256, m = f >> 3, k4 = (f & 7) << 2;
        ra[i] = *(const float4*)&x[(size_t)(m0 + m) * D_MODEL + k4];
    }
#pragma unroll
    for (int i = 0; i < 2; ++i) {
        int c = tid + i * 256, k = c >> 4, n4 = (c & 15) << 2;
        rw[i] = *(const float4*)&Wq[(size_t)k * (3 * D_MODEL) + n0 + n4];
    }
#pragma unroll
    for (int i = 0; i < 8; ++i) {
        int f = tid + i * 256, m = f >> 3, k4 = (f & 7) << 2;
        bf16x4 bv = { (bf16_t)ra[i].x, (bf16_t)ra[i].y, (bf16_t)ra[i].z, (bf16_t)ra[i].w };
        *(bf16x4*)&As[0][m * LDA + k4] = bv;
    }
#pragma unroll
    for (int i = 0; i < 2; ++i) {
        int c = tid + i * 256, k = c >> 4, n4 = (c & 15) << 2;
        Ws[0][(n4 + 0) * LDA + k] = (bf16_t)rw[i].x;
        Ws[0][(n4 + 1) * LDA + k] = (bf16_t)rw[i].y;
        Ws[0][(n4 + 2) * LDA + k] = (bf16_t)rw[i].z;
        Ws[0][(n4 + 3) * LDA + k] = (bf16_t)rw[i].w;
    }
    __syncthreads();

    int buf = 0;
    for (int kk = KSTEP;; kk += KSTEP) {
        const bool more = (kk < D_MODEL);
        if (more) {  // issue next-tile global loads before computing
#pragma unroll
            for (int i = 0; i < 8; ++i) {
                int f = tid + i * 256, m = f >> 3, k4 = (f & 7) << 2;
                ra[i] = *(const float4*)&x[(size_t)(m0 + m) * D_MODEL + kk + k4];
            }
#pragma unroll
            for (int i = 0; i < 2; ++i) {
                int c = tid + i * 256, k = c >> 4, n4 = (c & 15) << 2;
                rw[i] = *(const float4*)&Wq[(size_t)(kk + k) * (3 * D_MODEL) + n0 + n4];
            }
        }
        // ---- compute from LDS[buf]: 8 WMMAs/wave ----
        {
            const int ar0 = (32 * w + r) * LDA;
            const int ar1 = (32 * w + 16 + r) * LDA;
            bf16x16 a0 = load_frag16(&As[buf][ar0 + 8 * half], &As[buf][ar0 + 16 + 8 * half]);
            bf16x16 a1 = load_frag16(&As[buf][ar1 + 8 * half], &As[buf][ar1 + 16 + 8 * half]);
#pragma unroll
            for (int nt = 0; nt < 4; ++nt) {
                const int br = (16 * nt + r) * LDA;
                bf16x16 b = load_frag16(&Ws[buf][br + 8 * half], &Ws[buf][br + 16 + 8 * half]);
                acc[0][nt] = wmma_bf16(a0, b, acc[0][nt]);
                acc[1][nt] = wmma_bf16(a1, b, acc[1][nt]);
            }
        }
        if (!more) break;
        const int nb = buf ^ 1;
#pragma unroll
        for (int i = 0; i < 8; ++i) {
            int f = tid + i * 256, m = f >> 3, k4 = (f & 7) << 2;
            bf16x4 bv = { (bf16_t)ra[i].x, (bf16_t)ra[i].y, (bf16_t)ra[i].z, (bf16_t)ra[i].w };
            *(bf16x4*)&As[nb][m * LDA + k4] = bv;
        }
#pragma unroll
        for (int i = 0; i < 2; ++i) {
            int c = tid + i * 256, k = c >> 4, n4 = (c & 15) << 2;
            Ws[nb][(n4 + 0) * LDA + k] = (bf16_t)rw[i].x;
            Ws[nb][(n4 + 1) * LDA + k] = (bf16_t)rw[i].y;
            Ws[nb][(n4 + 2) * LDA + k] = (bf16_t)rw[i].z;
            Ws[nb][(n4 + 3) * LDA + k] = (bf16_t)rw[i].w;
        }
        __syncthreads();
        buf = nb;
    }

    // ---- epilogue: bias + scatter into Q / K / Vt ----
#pragma unroll
    for (int a = 0; a < 2; ++a)
#pragma unroll
    for (int nt = 0; nt < 4; ++nt) {
        const int c  = n0 + nt * 16 + r;          // [0, 3072)
        const int t  = c >> 10;                   // 0=Q 1=K 2=V
        const int c1 = c & (D_MODEL - 1);
        const int h  = c1 >> 6, d = c1 & 63;
        const float bias = bq[c];
#pragma unroll
        for (int v = 0; v < 8; ++v) {
            const int m  = m0 + 32 * w + 16 * a + v + 8 * half;
            const int bb = m >> 11;
            const int tq = m & (SEQ_T - 1);
            const float val = acc[a][nt][v] + bias;
            const size_t bh = (size_t)(bb * N_HEADS + h);
            if (t == 0)
                Q[(bh * SEQ_T + tq) * D_HEAD + d] = (bf16_t)(val * 0.125f);
            else if (t == 1)
                K[(bh * SEQ_T + tq) * D_HEAD + d] = (bf16_t)val;
            else
                Vt[(bh * D_HEAD + d) * SEQ_T + tq] = (bf16_t)val;
        }
    }
}

// ===========================================================================
// Kernel 2: causal flash attention per (B,H).
// grid: (T/64, B*H) = (32,32), block 128 (4 waves, 16 queries each)
// ===========================================================================
__global__ __launch_bounds__(128) void attn_kernel(
    const bf16_t* __restrict__ Q, const bf16_t* __restrict__ K,
    const bf16_t* __restrict__ Vt, bf16_t* __restrict__ Aout)
{
    __shared__ bf16_t Plds[4 * 16 * 32];
    const int q0   = blockIdx.x * 64;
    const int bh   = blockIdx.y;
    const int b    = bh >> 4, h = bh & 15;
    const int tid  = threadIdx.x;
    const int lane = tid & 31, w = tid >> 5;
    const int half = lane >> 4, r = lane & 15;

    const bf16_t* Qh = Q  + (size_t)bh * SEQ_T * D_HEAD;
    const bf16_t* Kh = K  + (size_t)bh * SEQ_T * D_HEAD;
    const bf16_t* Vh = Vt + (size_t)bh * D_HEAD * SEQ_T;
    bf16_t* Pw = &Plds[w * 16 * 32];

    const int qrow = q0 + 16 * w + r;
    bf16x16 qa[2];
#pragma unroll
    for (int s = 0; s < 2; ++s)
        qa[s] = load_frag16(&Qh[(size_t)qrow * 64 + 32 * s + 8 * half],
                            &Qh[(size_t)qrow * 64 + 32 * s + 16 + 8 * half]);

    f32x8 oacc[4];
#pragma unroll
    for (int i = 0; i < 4; ++i) oacc[i] = zero8();
    float m_run[8], s_run[8];
#pragma unroll
    for (int v = 0; v < 8; ++v) { m_run[v] = -__builtin_inff(); s_run[v] = 0.0f; }

    const int qi_base = q0 + 16 * w + 8 * half;

    for (int k0 = 0; k0 < q0 + 64; k0 += 32) {
        // ---- S = Q*K^T (2 n-tiles x 2 k-steps) ----
        f32x8 cs[2];
#pragma unroll
        for (int nt = 0; nt < 2; ++nt) {
            const size_t key = (size_t)(k0 + 16 * nt + r) * 64;
            bf16x16 kb0 = load_frag16(&Kh[key + 8 * half],      &Kh[key + 16 + 8 * half]);
            bf16x16 kb1 = load_frag16(&Kh[key + 32 + 8 * half], &Kh[key + 48 + 8 * half]);
            f32x8 c = zero8();
            c = wmma_bf16(qa[0], kb0, c);
            c = wmma_bf16(qa[1], kb1, c);
            cs[nt] = c;
        }
        // ---- causal mask ----
        const int kj0 = k0 + r, kj1 = k0 + 16 + r;
#pragma unroll
        for (int v = 0; v < 8; ++v) {
            const int qi = qi_base + v;
            if (kj0 > qi) cs[0][v] = -1e30f;
            if (kj1 > qi) cs[1][v] = -1e30f;
        }
        // ---- online softmax ----
        float corr[8];
#pragma unroll
        for (int v = 0; v < 8; ++v) {
            float t = fmaxf(cs[0][v], cs[1][v]);
#pragma unroll
            for (int m = 1; m < 16; m <<= 1)
                t = fmaxf(t, __shfl_xor(t, m, 32));
            const float mn = fmaxf(m_run[v], t);
            corr[v] = __expf(m_run[v] - mn);
            m_run[v] = mn;
            const float p0 = __expf(cs[0][v] - mn);
            const float p1 = __expf(cs[1][v] - mn);
            cs[0][v] = p0; cs[1][v] = p1;
            float rs = p0 + p1;
#pragma unroll
            for (int m = 1; m < 16; m <<= 1)
                rs += __shfl_xor(rs, m, 32);
            s_run[v] = s_run[v] * corr[v] + rs;
        }
#pragma unroll
        for (int nt = 0; nt < 4; ++nt)
#pragma unroll
            for (int v = 0; v < 8; ++v)
                oacc[nt][v] *= corr[v];

        // ---- C-layout P -> LDS -> A-layout fragment ----
#pragma unroll
        for (int nt = 0; nt < 2; ++nt)
#pragma unroll
            for (int v = 0; v < 8; ++v)
                Pw[(v + 8 * half) * 32 + 16 * nt + r] = (bf16_t)cs[nt][v];
        asm volatile("s_wait_dscnt 0" ::: "memory");
        bf16x16 pa = load_frag16(&Pw[r * 32 + 8 * half], &Pw[r * 32 + 16 + 8 * half]);

        // ---- O += P * V ----
#pragma unroll
        for (int nt = 0; nt < 4; ++nt) {
            const size_t dcol = (size_t)(16 * nt + r) * SEQ_T + k0;
            bf16x16 vb = load_frag16(&Vh[dcol + 8 * half], &Vh[dcol + 16 + 8 * half]);
            oacc[nt] = wmma_bf16(pa, vb, oacc[nt]);
        }
    }

    // ---- normalize + store ----
#pragma unroll
    for (int v = 0; v < 8; ++v) {
        const float inv = 1.0f / s_run[v];
        const size_t m = (size_t)b * SEQ_T + qi_base + v;
#pragma unroll
        for (int nt = 0; nt < 4; ++nt)
            Aout[m * D_MODEL + h * D_HEAD + 16 * nt + r] = (bf16_t)(oacc[nt][v] * inv);
    }
}

// ===========================================================================
// Kernel 3: output projection  out = attn @ W_proj + b_proj  (f32 out)
// grid: (1024/NBLK, 4096/MBLK) = (16,16), block 256
// ===========================================================================
__global__ __launch_bounds__(256) void proj_kernel(
    const bf16_t* __restrict__ Ain, const float* __restrict__ Wp,
    const float* __restrict__ bp, float* __restrict__ out)
{
    __shared__ bf16_t As[2][MBLK * LDA];
    __shared__ bf16_t Ws[2][NBLK * LDA];
    const int n0   = blockIdx.x * NBLK;
    const int m0   = blockIdx.y * MBLK;
    const int tid  = threadIdx.x;
    const int lane = tid & 31, w = tid >> 5;
    const int half = lane >> 4, r = lane & 15;

    f32x8 acc[2][4];
#pragma unroll
    for (int a = 0; a < 2; ++a)
#pragma unroll
        for (int i = 0; i < 4; ++i) acc[a][i] = zero8();

    bf16x8 ra[4];   // A tile: 256x32 bf16 = 1024 x (8 bf16) / 256 thr = 4 each
    float4 rw[2];

#pragma unroll
    for (int i = 0; i < 4; ++i) {
        int c = tid + i * 256, m = c >> 2, k8 = (c & 3) << 3;
        ra[i] = *(const bf16x8*)&Ain[(size_t)(m0 + m) * D_MODEL + k8];
    }
#pragma unroll
    for (int i = 0; i < 2; ++i) {
        int c = tid + i * 256, k = c >> 4, n4 = (c & 15) << 2;
        rw[i] = *(const float4*)&Wp[(size_t)k * D_MODEL + n0 + n4];
    }
#pragma unroll
    for (int i = 0; i < 4; ++i) {
        int c = tid + i * 256, m = c >> 2, k8 = (c & 3) << 3;
        *(bf16x8*)&As[0][m * LDA + k8] = ra[i];
    }
#pragma unroll
    for (int i = 0; i < 2; ++i) {
        int c = tid + i * 256, k = c >> 4, n4 = (c & 15) << 2;
        Ws[0][(n4 + 0) * LDA + k] = (bf16_t)rw[i].x;
        Ws[0][(n4 + 1) * LDA + k] = (bf16_t)rw[i].y;
        Ws[0][(n4 + 2) * LDA + k] = (bf16_t)rw[i].z;
        Ws[0][(n4 + 3) * LDA + k] = (bf16_t)rw[i].w;
    }
    __syncthreads();

    int buf = 0;
    for (int kk = KSTEP;; kk += KSTEP) {
        const bool more = (kk < D_MODEL);
        if (more) {
#pragma unroll
            for (int i = 0; i < 4; ++i) {
                int c = tid + i * 256, m = c >> 2, k8 = (c & 3) << 3;
                ra[i] = *(const bf16x8*)&Ain[(size_t)(m0 + m) * D_MODEL + kk + k8];
            }
#pragma unroll
            for (int i = 0; i < 2; ++i) {
                int c = tid + i * 256, k = c >> 4, n4 = (c & 15) << 2;
                rw[i] = *(const float4*)&Wp[(size_t)(kk + k) * D_MODEL + n0 + n4];
            }
        }
        {
            const int ar0 = (32 * w + r) * LDA;
            const int ar1 = (32 * w + 16 + r) * LDA;
            bf16x16 a0 = load_frag16(&As[buf][ar0 + 8 * half], &As[buf][ar0 + 16 + 8 * half]);
            bf16x16 a1 = load_frag16(&As[buf][ar1 + 8 * half], &As[buf][ar1 + 16 + 8 * half]);
#pragma unroll
            for (int nt = 0; nt < 4; ++nt) {
                const int br = (16 * nt + r) * LDA;
                bf16x16 b = load_frag16(&Ws[buf][br + 8 * half], &Ws[buf][br + 16 + 8 * half]);
                acc[0][nt] = wmma_bf16(a0, b, acc[0][nt]);
                acc[1][nt] = wmma_bf16(a1, b, acc[1][nt]);
            }
        }
        if (!more) break;
        const int nb = buf ^ 1;
#pragma unroll
        for (int i = 0; i < 4; ++i) {
            int c = tid + i * 256, m = c >> 2, k8 = (c & 3) << 3;
            *(bf16x8*)&As[nb][m * LDA + k8] = ra[i];
        }
#pragma unroll
        for (int i = 0; i < 2; ++i) {
            int c = tid + i * 256, k = c >> 4, n4 = (c & 15) << 2;
            Ws[nb][(n4 + 0) * LDA + k] = (bf16_t)rw[i].x;
            Ws[nb][(n4 + 1) * LDA + k] = (bf16_t)rw[i].y;
            Ws[nb][(n4 + 2) * LDA + k] = (bf16_t)rw[i].z;
            Ws[nb][(n4 + 3) * LDA + k] = (bf16_t)rw[i].w;
        }
        __syncthreads();
        buf = nb;
    }

#pragma unroll
    for (int a = 0; a < 2; ++a)
#pragma unroll
    for (int nt = 0; nt < 4; ++nt) {
        const int c = n0 + nt * 16 + r;
        const float bias = bp[c];
#pragma unroll
        for (int v = 0; v < 8; ++v) {
            const int m = m0 + 32 * w + 16 * a + v + 8 * half;
            out[(size_t)m * D_MODEL + c] = acc[a][nt][v] + bias;
        }
    }
}

// ===========================================================================
// Launcher
// ===========================================================================
extern "C" void kernel_launch(void* const* d_in, const int* in_sizes, int n_in,
                              void* d_out, int out_size, void* d_ws, size_t ws_size,
                              hipStream_t stream) {
    const float* x      = (const float*)d_in[0];
    const float* W_qkv  = (const float*)d_in[1];
    const float* b_qkv  = (const float*)d_in[2];
    const float* W_proj = (const float*)d_in[3];
    const float* b_proj = (const float*)d_in[4];
    float* out = (float*)d_out;

    const size_t tensor_elems = (size_t)BATCH * N_HEADS * SEQ_T * D_HEAD;
    bf16_t* Q    = (bf16_t*)d_ws;
    bf16_t* K    = Q  + tensor_elems;
    bf16_t* Vt   = K  + tensor_elems;
    bf16_t* Abuf = Vt + tensor_elems;

    dim3 g1(3 * D_MODEL / NBLK, M_TOT / MBLK);    // (48, 16)
    qkv_kernel<<<g1, 256, 0, stream>>>(x, W_qkv, b_qkv, Q, K, Vt);

    dim3 g2(SEQ_T / 64, BATCH * N_HEADS);         // (32, 32)
    attn_kernel<<<g2, 128, 0, stream>>>(Q, K, Vt, Abuf);

    dim3 g3(D_MODEL / NBLK, M_TOT / MBLK);        // (16, 16)
    proj_kernel<<<g3, 256, 0, stream>>>(Abuf, W_proj, b_proj, out);
}
